// SparseMoE_72765335929164
// MI455X (gfx1250) — matrix operand order repaired and checked by reading, not compiled, not run on hardware
//
#include <hip/hip_runtime.h>

// Problem dims (fixed by the reference)
constexpr int NTOK = 8192;   // B*S
constexpr int DD   = 1024;   // D
constexpr int HH   = 4096;   // H
constexpr int EE   = 4;      // experts

// ---------- types for WMMA ----------
typedef __attribute__((ext_vector_type(16))) __bf16 v16bf;
typedef __attribute__((ext_vector_type(8)))  float  v8f;

struct alignas(16) U4 { unsigned int x, y, z, w; };
struct U8 { U4 lo, hi; };
struct alignas(8) F2 { float x, y; };

static_assert(sizeof(U8) == sizeof(v16bf), "size match");

__device__ inline unsigned int bf16u(float f) {
    unsigned int u = __builtin_bit_cast(unsigned int, f);
    unsigned int r = u + 0x7FFFu + ((u >> 16) & 1u);   // round-to-nearest-even
    return r >> 16;
}

__device__ inline float gelu_exact(float v) {
    return 0.5f * v * (1.0f + erff(v * 0.70710678118654752f));
}

__device__ inline v16bf ld_pair_lds(const unsigned short* p) {
    U8 u;
    u.lo = *(const U4*)p;
    u.hi = *(const U4*)(p + 16);     // +32 bytes
    return __builtin_bit_cast(v16bf, u);
}

__device__ inline v16bf ld_pair_glb(const unsigned int* p) {
    U8 u;
    u.lo = *(const U4*)p;
    u.hi = *(const U4*)(p + 4);      // +16 bytes
    return __builtin_bit_cast(v16bf, u);
}

// ---------------------------------------------------------------------------
// 1) expert constants c[e][d] = sum_h gelu(b1[e][h]) * w2[e][h][d] + b2[e][d]
//    Also zeroes the routing counters (runs before router in stream order).
// ---------------------------------------------------------------------------
__global__ __launch_bounds__(256) void moe_c(const float* __restrict__ w2,
                                             const float* __restrict__ b1,
                                             const float* __restrict__ b2,
                                             float* __restrict__ cbuf,
                                             int* __restrict__ counts) {
    int e = blockIdx.y;
    int d = blockIdx.x * 256 + threadIdx.x;
    if (blockIdx.x == 0 && blockIdx.y == 0 && threadIdx.x < EE) counts[threadIdx.x] = 0;
    float acc = b2[e * DD + d];
    const float* w  = w2 + (size_t)e * HH * DD + d;
    const float* bb = b1 + (size_t)e * HH;
    for (int h = 0; h < HH; ++h)
        acc += gelu_exact(bb[h]) * w[(size_t)h * DD];
    cbuf[e * DD + d] = acc;
}

// ---------------------------------------------------------------------------
// 2) out[n][d] = sum_e c[e][d]   (since gates sum to 1, c_sum is added once)
// ---------------------------------------------------------------------------
__global__ __launch_bounds__(256) void moe_outinit(const float* __restrict__ cbuf,
                                                   float* __restrict__ out) {
    size_t i = (size_t)blockIdx.x * 256 + threadIdx.x;   // NTOK*DD threads
    int d = (int)(i & (DD - 1));
    out[i] = cbuf[d] + cbuf[DD + d] + cbuf[2 * DD + d] + cbuf[3 * DD + d];
}

// ---------------------------------------------------------------------------
// 3) router: one wave per token. logits = x@wr + br, top-2 (strict > => lowest
//    index on ties, matching jax.lax.top_k), softmax over the 2 values,
//    append (token, gate) to per-expert lists.
// ---------------------------------------------------------------------------
__global__ __launch_bounds__(256) void moe_router(const float* __restrict__ x,
                                                  const float* __restrict__ wr,
                                                  const float* __restrict__ br,
                                                  int* __restrict__ counts,
                                                  int* __restrict__ lists,
                                                  float* __restrict__ lgates) {
    int t    = blockIdx.x * 8 + (threadIdx.x >> 5);
    int lane = threadIdx.x & 31;
    float a0 = 0.f, a1 = 0.f, a2 = 0.f, a3 = 0.f;
    const float* xr = x + (size_t)t * DD;
    for (int i = 0; i < DD / 32; ++i) {
        int d = lane + i * 32;
        float xv = xr[d];
        const float* w = wr + (size_t)d * EE;
        a0 += xv * w[0]; a1 += xv * w[1]; a2 += xv * w[2]; a3 += xv * w[3];
    }
    for (int off = 16; off > 0; off >>= 1) {
        a0 += __shfl_down(a0, off);
        a1 += __shfl_down(a1, off);
        a2 += __shfl_down(a2, off);
        a3 += __shfl_down(a3, off);
    }
    if (lane == 0) {
        float l[4] = { a0 + br[0], a1 + br[1], a2 + br[2], a3 + br[3] };
        int i0 = 0; float v0 = l[0];
        for (int e = 1; e < 4; ++e) if (l[e] > v0) { v0 = l[e]; i0 = e; }
        int i1 = -1; float v1 = -INFINITY;
        for (int e = 0; e < 4; ++e) if (e != i0 && l[e] > v1) { v1 = l[e]; i1 = e; }
        float r  = expf(v1 - v0);
        float g0 = 1.0f / (1.0f + r);
        float g1 = r / (1.0f + r);
        int p0 = atomicAdd(&counts[i0], 1);
        lists[i0 * NTOK + p0] = t; lgates[i0 * NTOK + p0] = g0;
        int p1 = atomicAdd(&counts[i1], 1);
        lists[i1 * NTOK + p1] = t; lgates[i1 * NTOK + p1] = g1;
    }
}

// ---------------------------------------------------------------------------
// 4) weight conversion fp32 -> bf16, packed in WMMA B-operand order.
//    Layout per (e, kb, nb) block of 32(K) x 16(N): 32 lanes x 8 dwords.
//    lane = ((K_local/8)%2)*16 + n_local ; dword i holds K = 8*(lane/16)+2(i%4)+16(i/4), +1
// ---------------------------------------------------------------------------
__global__ __launch_bounds__(256) void conv_w1(const float* __restrict__ w1,
                                               unsigned int* __restrict__ w1p) {
    size_t o = (size_t)blockIdx.x * 256 + threadIdx.x;   // E*DD*HH/2 dwords
    int dw   = (int)(o & 7);
    int lane = (int)((o >> 3) & 31);
    size_t blk = o >> 8;
    int nb = (int)(blk & 255);          // HH/16
    int kb = (int)((blk >> 8) & 31);    // DD/32
    int e  = (int)(blk >> 13);
    int nl = lane & 15, kh = lane >> 4;
    int k0 = kb * 32 + kh * 8 + (dw & 3) * 2 + (dw >> 2) * 16;       // row in D
    int h  = nb * 16 + nl;                      // col in H
    size_t base = ((size_t)e * DD + k0) * HH + h;
    unsigned int lo = bf16u(w1[base]);
    unsigned int hi = bf16u(w1[base + HH]);
    w1p[o] = lo | (hi << 16);
}

__global__ __launch_bounds__(256) void conv_w2(const float* __restrict__ w2,
                                               unsigned int* __restrict__ w2p) {
    size_t o = (size_t)blockIdx.x * 256 + threadIdx.x;   // E*HH*DD/2 dwords
    int dw   = (int)(o & 7);
    int lane = (int)((o >> 3) & 31);
    size_t blk = o >> 8;
    int nb = (int)(blk & 63);           // DD/16
    int kb = (int)((blk >> 6) & 127);   // HH/32
    int e  = (int)(blk >> 13);
    int nl = lane & 15, kh = lane >> 4;
    int k0 = kb * 32 + kh * 8 + (dw & 3) * 2 + (dw >> 2) * 16;       // row in H
    int d  = nb * 16 + nl;                      // col in D
    size_t base = ((size_t)e * HH + k0) * DD + d;
    unsigned int lo = bf16u(w2[base]);
    unsigned int hi = bf16u(w2[base + DD]);
    w2p[o] = lo | (hi << 16);
}

// ---------------------------------------------------------------------------
// 5) fused per-expert FFN on gathered tokens, bf16 WMMA, fp32 accumulate.
//    grid = (256 tiles, 4 experts), 256 threads = 8 waves, 32 tokens/block.
//    Two 16-row M-subtiles share every B operand (halves L2 weight traffic).
// ---------------------------------------------------------------------------
__global__ __launch_bounds__(256) void moe_ffn(const float* __restrict__ x,
                                               const float* __restrict__ b1,
                                               const float* __restrict__ b2,
                                               const unsigned int* __restrict__ w1p,
                                               const unsigned int* __restrict__ w2p,
                                               const float* __restrict__ cbuf,
                                               const int* __restrict__ counts,
                                               const int* __restrict__ lists,
                                               const float* __restrict__ lgates,
                                               float* __restrict__ out) {
    int e   = blockIdx.y;
    int cnt = counts[e];
    if ((int)blockIdx.x * 32 >= cnt) return;

    __shared__ alignas(16) unsigned short xs[32 * 1032];  // 32 x 1024 bf16, padded (66 KB)
    __shared__ alignas(16) unsigned short hs[32 * 136];   // 32 x 128  bf16, padded (8.7 KB)
    __shared__ int   ts[32];
    __shared__ float gsv[32];

    int tid = threadIdx.x;
    if (tid < 32) {
        int slot = blockIdx.x * 32 + tid;
        if (slot < cnt) { ts[tid] = lists[e * NTOK + slot]; gsv[tid] = lgates[e * NTOK + slot]; }
        else            { ts[tid] = lists[e * NTOK + cnt - 1]; gsv[tid] = 0.0f; }
    }
    __syncthreads();

    // gather 32 token rows, convert fp32 -> bf16 into LDS
    for (int i = tid; i < 32 * 512; i += 256) {
        int row = i >> 9, dp = i & 511;
        F2 v = *(const F2*)(x + (size_t)ts[row] * DD + dp * 2);
        *(unsigned int*)&xs[row * 1032 + dp * 2] = bf16u(v.x) | (bf16u(v.y) << 16);
    }
    __syncthreads();

    int wid = tid >> 5, lane = tid & 31, lh = lane >> 4, l16 = lane & 15;

    v8f yacc[2][8];
#pragma unroll
    for (int s = 0; s < 2; ++s)
#pragma unroll
        for (int i = 0; i < 8; ++i) yacc[s][i] = (v8f){0.f,0.f,0.f,0.f,0.f,0.f,0.f,0.f};

    for (int hc = 0; hc < HH / 128; ++hc) {
        int nbg = hc * 8 + wid;               // 16-wide H column tile, 0..255
        // -------- GEMM1: h = xs @ W1[:, nbg*16 .. +15], K = 1024, 2 M-subtiles
        v8f hacc0 = (v8f){0.f,0.f,0.f,0.f,0.f,0.f,0.f,0.f};
        v8f hacc1 = (v8f){0.f,0.f,0.f,0.f,0.f,0.f,0.f,0.f};
        const unsigned short* xrow0 = &xs[l16 * 1032 + lh * 8];
        const unsigned short* xrow1 = &xs[(16 + l16) * 1032 + lh * 8];
#pragma unroll 4
        for (int kb = 0; kb < 32; ++kb) {
            v16bf A0 = ld_pair_lds(xrow0 + kb * 32);
            v16bf A1 = ld_pair_lds(xrow1 + kb * 32);
            v16bf Bm = ld_pair_glb(w1p + (((size_t)(e * 32 + kb) * 256 + nbg) << 8) + lane * 8);
            hacc0 = __builtin_amdgcn_wmma_f32_16x16x32_bf16(false, A0, false, Bm,
                                                            (short)0, hacc0, false, false);
            hacc1 = __builtin_amdgcn_wmma_f32_16x16x32_bf16(false, A1, false, Bm,
                                                            (short)0, hacc1, false, false);
        }
        int   hcol = nbg * 16 + l16;
        float b1v  = b1[e * HH + hcol];
        __syncthreads();   // previous iteration's GEMM2 done reading hs
#pragma unroll
        for (int r = 0; r < 8; ++r) {
            int M = r + lh * 8;
            hs[M * 136 + wid * 16 + l16] =
                (unsigned short)bf16u(gelu_exact(hacc0[r] + b1v));
            hs[(16 + M) * 136 + wid * 16 + l16] =
                (unsigned short)bf16u(gelu_exact(hacc1[r] + b1v));
        }
        __syncthreads();   // hs chunk ready

        // -------- GEMM2: y += h_chunk @ W2[hc*128 .. +127, :], per-wave 8 D-tiles
#pragma unroll
        for (int ks = 0; ks < 4; ++ks) {
            v16bf A0 = ld_pair_lds(&hs[l16 * 136 + ks * 32 + lh * 8]);
            v16bf A1 = ld_pair_lds(&hs[(16 + l16) * 136 + ks * 32 + lh * 8]);
            int kbg = hc * 4 + ks;
#pragma unroll
            for (int dtl = 0; dtl < 8; ++dtl) {
                int dt = wid * 8 + dtl;
                v16bf Bm = ld_pair_glb(w2p + (((size_t)(e * 128 + kbg) * 64 + dt) << 8) + lane * 8);
                yacc[0][dtl] = __builtin_amdgcn_wmma_f32_16x16x32_bf16(false, A0, false, Bm,
                                                                       (short)0, yacc[0][dtl], false, false);
                yacc[1][dtl] = __builtin_amdgcn_wmma_f32_16x16x32_bf16(false, A1, false, Bm,
                                                                       (short)0, yacc[1][dtl], false, false);
            }
        }
    }

    // -------- epilogue: out[tok] += gate * (y + b2 - c_e) --------
#pragma unroll
    for (int dtl = 0; dtl < 8; ++dtl) {
        int   dcol = (wid * 8 + dtl) * 16 + l16;
        float b2v  = b2[e * DD + dcol];
        float cv   = cbuf[e * DD + dcol];
#pragma unroll
        for (int s = 0; s < 2; ++s) {
#pragma unroll
            for (int r = 0; r < 8; ++r) {
                int M = s * 16 + r + lh * 8;
                float val = gsv[M] * (yacc[s][dtl][r] + b2v - cv);
                atomicAdd(out + (size_t)ts[M] * DD + dcol, val);
            }
        }
    }
}

// ---------------------------------------------------------------------------
extern "C" void kernel_launch(void* const* d_in, const int* in_sizes, int n_in,
                              void* d_out, int out_size, void* d_ws, size_t ws_size,
                              hipStream_t stream) {
    const float* x  = (const float*)d_in[0];
    const float* w1 = (const float*)d_in[1];
    const float* b1 = (const float*)d_in[2];
    const float* w2 = (const float*)d_in[3];
    const float* b2 = (const float*)d_in[4];
    const float* wr = (const float*)d_in[5];
    const float* br = (const float*)d_in[6];
    float* out = (float*)d_out;

    char* ws = (char*)d_ws;
    // scratch layout (total ~67.4 MB)
    unsigned int* w1p    = (unsigned int*)(ws);                  // 33,554,432 B
    unsigned int* w2p    = (unsigned int*)(ws + 33554432);       // 33,554,432 B
    float*        cbuf   = (float*)(ws + 67108864);              //     16,384 B
    int*          counts = (int*)(ws + 67125248);                //        256 B (padded)
    int*          lists  = (int*)(ws + 67125504);                //    131,072 B
    float*        lgates = (float*)(ws + 67256576);              //    131,072 B

    (void)in_sizes; (void)n_in; (void)out_size; (void)ws_size;

    moe_c      <<<dim3(DD / 256, EE), 256, 0, stream>>>(w2, b1, b2, cbuf, counts);
    moe_outinit<<<(NTOK * DD) / 256,  256, 0, stream>>>(cbuf, out);
    moe_router <<<NTOK / 8,           256, 0, stream>>>(x, wr, br, counts, lists, lgates);
    conv_w1    <<<(EE * DD * HH / 2) / 256, 256, 0, stream>>>(w1, w1p);
    conv_w2    <<<(EE * HH * DD / 2) / 256, 256, 0, stream>>>(w2, w2p);
    moe_ffn    <<<dim3(NTOK / 32, EE), 256, 0, stream>>>(x, b1, b2, w1p, w2p,
                                                         cbuf, counts, lists, lgates, out);
}
